// EquivariantBlock_70317204570116
// MI455X (gfx1250) — matrix operand order repaired and checked by reading, not compiled, not run on hardware
//
#include <hip/hip_runtime.h>
#include <hip/hip_bf16.h>
#include <math.h>

// ---------------- problem constants (match reference) ----------------
#define MUL0      32
#define MUL1      16
#define HID       64
#define EDGE_DIM  32
#define WNUM      2304           // 32*32 + 16*32 + 32*16 + 16*16
#define FDIM      80             // MUL0 + 3*MUL1
#define EPS_F     1e-5f
#define ALPHA_F   0.14433756729740643f   // 1/sqrt(48)
#define INV_SQRT3 0.57735026918962576f

typedef __attribute__((ext_vector_type(16))) __bf16 v16bf;
typedef __attribute__((ext_vector_type(8)))  float  v8f;

__device__ __forceinline__ v8f wmma_bf16(v16bf a, v16bf b, v8f c) {
  // D = A(16x32 bf16) x B(32x16 bf16) + C(16x16 f32)
  return __builtin_amdgcn_wmma_f32_16x16x32_bf16(false, a, false, b,
                                                 (short)0, c, false, false);
}

__device__ __forceinline__ v8f splat8(float x) {
  v8f v;
  #pragma unroll
  for (int r = 0; r < 8; ++r) v[r] = x;
  return v;
}

// wave-local LDS producer->consumer hand-off (all traffic within one wave32).
__device__ __forceinline__ void wave_lds_fence() {
  asm volatile("s_wait_dscnt 0" ::: "memory");
  __builtin_amdgcn_wave_barrier();
}

// v_rcp_f32-based sigmoid: avoids the v_div_scale/v_div_fmas divide sequence;
// exp and rcp are TRANS ops that co-issue with regular VALU.
__device__ __forceinline__ float fast_sigmoid(float x) {
  return __builtin_amdgcn_rcpf(1.0f + __expf(-x));
}
__device__ __forceinline__ float silu_f(float x) { return x * fast_sigmoid(x); }

// =====================================================================
// Kernel 0: zero scratch (scatter targets + stat accumulators)
// =====================================================================
__global__ void zero_kernel(float* __restrict__ p, size_t n) {
  size_t i = (size_t)blockIdx.x * blockDim.x + threadIdx.x;
  if (i < n) p[i] = 0.0f;
}

// =====================================================================
// Kernel P: pack a f32 weight matrix (K x ld, row-major) into bf16
// per-lane WMMA B-fragment order:
//   out[((tile*nchunks + kc)*32 + lane)*16 + j]
//     = bf16( W[(kc*32 + 8*(lane>>4) + (j&7) + ((j&8)<<1)) * ld + tile*16 + (lane&15)] )
// Hot-loop fragment load then becomes one contiguous 32 B vector per lane
// (global_load_b128 x2, lane-coalesced), no converts in the loop.
// =====================================================================
__global__ void pack_b_kernel(const float* __restrict__ W, __bf16* __restrict__ out,
                              int ld, int nchunks, size_t total) {
  size_t idx = (size_t)blockIdx.x * blockDim.x + threadIdx.x;
  if (idx >= total) return;
  int    j    = (int)(idx & 15);
  int    lane = (int)((idx >> 4) & 31);
  size_t tc   = idx >> 9;                 // combined (tile*nchunks + kc)
  int    kc   = (int)(tc % nchunks);
  int    K    = kc * 32 + 8 * (lane >> 4) + (j & 7) + ((j & 8) << 1);
  int    col  = (int)(tc / nchunks) * 16 + (lane & 15);
  out[idx] = (__bf16)W[(size_t)K * ld + col];
}

// =====================================================================
// Kernel 1: fused edge pipeline. One wave32 owns a 16-edge tile:
//   x1 = silu(ef@W1+b1), x2 = silu(x1@W2+b2),
//   w  = x2@W3+b3 (2304 cols): 144 tiles x 2 WMMAs (bias carried in the
//   WMMA C operand), each 16x16 tile of w consumed immediately by the
//   equivariant bilinear contraction in C-fragment layout (contraction
//   index i is constant per 16-col tile).
// Then atomic scatter of the 80-wide message into summed[dst].
// =====================================================================
__global__ __launch_bounds__(128)
void edge_kernel(const float* __restrict__ h,
                 const float* __restrict__ edge_sh,
                 const float* __restrict__ edge_features,
                 const int*   __restrict__ edge_index,
                 const __bf16* __restrict__ pW1, const float* __restrict__ b1,
                 const __bf16* __restrict__ pW2, const float* __restrict__ b2,
                 const __bf16* __restrict__ pW3, const float* __restrict__ b3,
                 float* __restrict__ summed, float* __restrict__ deg,
                 int E)
{
  // per-wave staging (4 waves / block)
  __shared__ float sA0 [4][16][32];      // se * sh0
  __shared__ float sSE [4][16][32];      // se
  __shared__ float sDV [4][16][16];      // dot(v, sh1)/sqrt(3)
  __shared__ float sVSH[4][16][16][3];   // v * sh0
  __shared__ float sSH1[4][16][3];       // sh1
  __shared__ float sX  [4][16][64];      // MLP activation staging (x1 then x2)

  const int lane = threadIdx.x & 31;
  const int w    = threadIdx.x >> 5;     // wave in block
  const int lh   = lane >> 4;            // lane half (fragment row group)
  const int nc   = lane & 15;            // column within 16-wide tile
  const int tile = blockIdx.x * 4 + w;
  const int base_e = tile * 16;
  if (base_e >= E) return;

  // packed fragment pointers for this lane (contiguous 32B per fragment)
  const __bf16* fW1 = pW1 + (size_t)lane * 16;
  const __bf16* fW2 = pW2 + (size_t)lane * 16;
  const __bf16* fW3 = pW3 + (size_t)lane * 16;

  // ---------------- stage per-edge equivariant operands ----------------
  for (int idx = lane; idx < 16 * 32; idx += 32) {
    int el = idx >> 5, i = idx & 31;
    int e  = base_e + el;
    float sv = 0.0f, s0 = 0.0f;
    if (e < E) {
      int sn = edge_index[e];                          // src node
      s0 = edge_sh[e * 4 + 0];
      sv = h[(size_t)sn * FDIM + i];
    }
    sSE[w][el][i] = sv;
    sA0[w][el][i] = sv * s0;
  }
  for (int idx = lane; idx < 16 * 16; idx += 32) {
    int el = idx >> 4, i = idx & 15;
    int e  = base_e + el;
    float vx=0, vy=0, vz=0, s0=0, sx=0, sy=0, sz=0;
    if (e < E) {
      int sn = edge_index[e];
      s0 = edge_sh[e*4+0]; sx = edge_sh[e*4+1];
      sy = edge_sh[e*4+2]; sz = edge_sh[e*4+3];
      const float* vp = h + (size_t)sn * FDIM + MUL0 + i * 3;
      vx = vp[0]; vy = vp[1]; vz = vp[2];
    }
    sDV [w][el][i]    = (vx*sx + vy*sy + vz*sz) * INV_SQRT3;
    sVSH[w][el][i][0] = vx * s0;
    sVSH[w][el][i][1] = vy * s0;
    sVSH[w][el][i][2] = vz * s0;
  }
  if (lane < 16) {
    int e = base_e + lane;
    sSH1[w][lane][0] = (e < E) ? edge_sh[e*4+1] : 0.0f;
    sSH1[w][lane][1] = (e < E) ? edge_sh[e*4+2] : 0.0f;
    sSH1[w][lane][2] = (e < E) ? edge_sh[e*4+3] : 0.0f;
  }

  // ---------------- layer 1: A fragment straight from global ----------
  v16bf a_ef;
  {
    int row = base_e + nc;
    const float* rp = edge_features + (size_t)min(row, E - 1) * EDGE_DIM;
    #pragma unroll
    for (int t = 0; t < 16; ++t) {
      int K = 8*lh + (t & 7) + ((t & 8) << 1);         // 16-bit A layout
      a_ef[t] = (__bf16)((row < E) ? rp[K] : 0.0f);
    }
  }
  #pragma unroll
  for (int n = 0; n < 4; ++n) {
    v16bf bf = *(const v16bf*)(fW1 + (size_t)n * 512);
    v8f c = splat8(b1[n*16 + nc]);       // bias rides the WMMA C operand
    c = wmma_bf16(a_ef, bf, c);
    #pragma unroll
    for (int r = 0; r < 8; ++r)
      sX[w][r + 8*lh][n*16 + nc] = silu_f(c[r]);
  }
  wave_lds_fence();

  // reload x1 as two K=32 A fragments (bf16 convert on the fly)
  v16bf a1lo, a1hi;
  {
    const float* rp = &sX[w][nc][0];
    #pragma unroll
    for (int t = 0; t < 16; ++t) {
      int K = 8*lh + (t & 7) + ((t & 8) << 1);
      a1lo[t] = (__bf16)rp[K];
      a1hi[t] = (__bf16)rp[32 + K];
    }
  }
  wave_lds_fence();   // all lanes done reading x1 before x2 overwrites sX

  // ---------------- layer 2 ----------------
  #pragma unroll
  for (int n = 0; n < 4; ++n) {
    v16bf bf0 = *(const v16bf*)(fW2 + (size_t)(n*2    ) * 512);
    v16bf bf1 = *(const v16bf*)(fW2 + (size_t)(n*2 + 1) * 512);
    v8f c = splat8(b2[n*16 + nc]);
    c = wmma_bf16(a1lo, bf0, c);
    c = wmma_bf16(a1hi, bf1, c);
    #pragma unroll
    for (int r = 0; r < 8; ++r)
      sX[w][r + 8*lh][n*16 + nc] = silu_f(c[r]);
  }
  wave_lds_fence();

  v16bf a2lo, a2hi;
  {
    const float* rp = &sX[w][nc][0];
    #pragma unroll
    for (int t = 0; t < 16; ++t) {
      int K = 8*lh + (t & 7) + ((t & 8) << 1);
      a2lo[t] = (__bf16)rp[K];
      a2hi[t] = (__bf16)rp[32 + K];
    }
  }

  // ---------------- layer 3 fused with tensor product -----------------
  v8f acc0lo = {}, acc0hi = {}, accP01 = {}, accV0 = {}, accV1 = {}, accV2 = {};

  #pragma unroll 2   // two independent load->WMMA->FMA chains in flight
  for (int tc = 0; tc < WNUM / 16; ++tc) {
    const int col0 = tc * 16;
    if (tc + 1 < WNUM / 16)   // global_prefetch_b8 on the packed stream
      __builtin_prefetch(fW3 + (size_t)(tc + 1) * 2 * 512, 0, 0);

    v16bf bf0 = *(const v16bf*)(fW3 + (size_t)(tc*2    ) * 512);
    v16bf bf1 = *(const v16bf*)(fW3 + (size_t)(tc*2 + 1) * 512);
    v8f c = splat8(b3[col0 + nc]);       // bias folded into accumulator
    c = wmma_bf16(a2lo, bf0, c);
    c = wmma_bf16(a2hi, bf1, c);

    if (col0 < 1024) {                       // w00: (i = col/32, o = col%32)
      const int i = col0 >> 5;
      if (((col0 >> 4) & 1) == 0) {
        #pragma unroll
        for (int r = 0; r < 8; ++r) acc0lo[r] += sA0[w][r + 8*lh][i] * c[r];
      } else {
        #pragma unroll
        for (int r = 0; r < 8; ++r) acc0hi[r] += sA0[w][r + 8*lh][i] * c[r];
      }
    } else if (col0 < 1536) {                // w10: dot(v,sh1) path
      const int l = col0 - 1024, i = l >> 5;
      if (((l >> 4) & 1) == 0) {
        #pragma unroll
        for (int r = 0; r < 8; ++r) acc0lo[r] += sDV[w][r + 8*lh][i] * c[r];
      } else {
        #pragma unroll
        for (int r = 0; r < 8; ++r) acc0hi[r] += sDV[w][r + 8*lh][i] * c[r];
      }
    } else if (col0 < 2048) {                // w01: se path (x sh1 later)
      const int i = (col0 - 1536) >> 4;
      #pragma unroll
      for (int r = 0; r < 8; ++r) accP01[r] += sSE[w][r + 8*lh][i] * c[r];
    } else {                                 // w11: v*sh0 path, 3 components
      const int i = (col0 - 2048) >> 4;
      #pragma unroll
      for (int r = 0; r < 8; ++r) {
        accV0[r] += sVSH[w][r + 8*lh][i][0] * c[r];
        accV1[r] += sVSH[w][r + 8*lh][i][1] * c[r];
        accV2[r] += sVSH[w][r + 8*lh][i][2] * c[r];
      }
    }
  }

  // ---------------- scatter 80-wide message into summed[dst] ----------
  #pragma unroll
  for (int r = 0; r < 8; ++r) {
    const int el = r + 8*lh;
    const int e  = base_e + el;
    if (e >= E) continue;
    const int dn = edge_index[E + e];                  // dst node
    float* srow = summed + (size_t)dn * FDIM;
    atomicAdd(&srow[nc],      ALPHA_F * acc0lo[r]);
    atomicAdd(&srow[16 + nc], ALPHA_F * acc0hi[r]);
    const float p01 = accP01[r];
    const float pv[3] = {accV0[r], accV1[r], accV2[r]};
    #pragma unroll
    for (int cc = 0; cc < 3; ++cc) {
      const float o1 = ALPHA_F * (p01 * sSH1[w][el][cc] + pv[cc]);
      atomicAdd(&srow[MUL0 + nc*3 + cc], o1);
    }
  }
  if (lane < 16) {
    const int e = base_e + lane;
    if (e < E) atomicAdd(&deg[edge_index[E + e]], 1.0f);
  }
}

// =====================================================================
// Kernel 2: agg + self-interaction; accumulate batch-norm statistics.
// stats[0:32]=sum_s, stats[32:64]=sumsq_s, stats[64:80]=sum mean_c(v^2)
// =====================================================================
__global__ __launch_bounds__(128)
void node_kernel1(const float* __restrict__ h,
                  const float* __restrict__ summed, const float* __restrict__ deg,
                  const float* __restrict__ ws_s, const float* __restrict__ bs_s,
                  const float* __restrict__ ws_v,
                  float* __restrict__ upd, float* __restrict__ stats, int N)
{
  __shared__ float red[80];
  const int n = blockIdx.x * blockDim.x + threadIdx.x;
  if (threadIdx.x < 80) red[threadIdx.x] = 0.0f;
  __syncthreads();

  if (n < N) {
    const float* hr = h + (size_t)n * FDIM;
    const float* sr = summed + (size_t)n * FDIM;
    float* ur = upd + (size_t)n * FDIM;
    const float inv = __builtin_amdgcn_rcpf(fmaxf(deg[n], 1.0f));

    float us[32];
    #pragma unroll 4
    for (int o = 0; o < 32; ++o) {
      float acc = bs_s[o];
      for (int i = 0; i < 32; ++i) acc += hr[i] * ws_s[i*32 + o];
      us[o] = sr[o] * inv + acc;
      ur[o] = us[o];
      atomicAdd(&red[o],      us[o]);
      atomicAdd(&red[32 + o], us[o] * us[o]);
    }
    #pragma unroll 4
    for (int o = 0; o < 16; ++o) {
      float a0 = 0, a1 = 0, a2 = 0;
      for (int i = 0; i < 16; ++i) {
        const float wv = ws_v[i*16 + o];
        a0 += hr[32 + i*3 + 0] * wv;
        a1 += hr[32 + i*3 + 1] * wv;
        a2 += hr[32 + i*3 + 2] * wv;
      }
      const float u0 = sr[32+o*3+0]*inv + a0;
      const float u1 = sr[32+o*3+1]*inv + a1;
      const float u2 = sr[32+o*3+2]*inv + a2;
      ur[32+o*3+0] = u0; ur[32+o*3+1] = u1; ur[32+o*3+2] = u2;
      atomicAdd(&red[64 + o], (u0*u0 + u1*u1 + u2*u2) * (1.0f/3.0f));
    }
  }
  __syncthreads();
  if (threadIdx.x < 80) atomicAdd(&stats[threadIdx.x], red[threadIdx.x]);
}

// =====================================================================
// Kernel 3: finalize statistics.
// =====================================================================
__global__ void stats_kernel(const float* __restrict__ stats,
                             float* __restrict__ norm, int N)
{
  const int f = threadIdx.x;
  const float invN = 1.0f / (float)N;
  if (f < 32) {
    const float mean = stats[f] * invN;
    const float var  = stats[32 + f] * invN - mean * mean;
    norm[f]      = mean;
    norm[32 + f] = rsqrtf(var + EPS_F);
  } else if (f < 48) {
    const int o = f - 32;
    norm[64 + o] = rsqrtf(stats[64 + o] * invN + EPS_F);
  }
}

// =====================================================================
// Kernel 4: normalize, gate, residual.
// =====================================================================
__global__ __launch_bounds__(128)
void node_kernel2(const float* __restrict__ h, const float* __restrict__ upd,
                  const float* __restrict__ norm,
                  const float* __restrict__ wg_s, const float* __restrict__ bg_s,
                  const float* __restrict__ wg_v,
                  const float* __restrict__ bn_ws, const float* __restrict__ bn_bs,
                  const float* __restrict__ bn_wv,
                  float* __restrict__ out, int N)
{
  const int n = blockIdx.x * blockDim.x + threadIdx.x;
  if (n >= N) return;
  const float* hr = h   + (size_t)n * FDIM;
  const float* ur = upd + (size_t)n * FDIM;
  float* orow = out + (size_t)n * FDIM;

  #pragma unroll 4
  for (int o = 0; o < 32; ++o) {
    float g = bg_s[o];
    for (int i = 0; i < 32; ++i) g += hr[i] * wg_s[i*32 + o];
    g = fast_sigmoid(g);
    const float ns = (ur[o] - norm[o]) * norm[32 + o] * bn_ws[o] + bn_bs[o];
    orow[o] = hr[o] + g * ns;
  }
  #pragma unroll 4
  for (int o = 0; o < 16; ++o) {
    float g0 = 0, g1 = 0, g2 = 0;
    for (int i = 0; i < 16; ++i) {
      const float wv = wg_v[i*16 + o];
      g0 += hr[32 + i*3 + 0] * wv;
      g1 += hr[32 + i*3 + 1] * wv;
      g2 += hr[32 + i*3 + 2] * wv;
    }
    const float sc = norm[64 + o] * bn_wv[o];
    orow[32+o*3+0] = hr[32+o*3+0] + fast_sigmoid(g0) * (ur[32+o*3+0] * sc);
    orow[32+o*3+1] = hr[32+o*3+1] + fast_sigmoid(g1) * (ur[32+o*3+1] * sc);
    orow[32+o*3+2] = hr[32+o*3+2] + fast_sigmoid(g2) * (ur[32+o*3+2] * sc);
  }
}

// =====================================================================
extern "C" void kernel_launch(void* const* d_in, const int* in_sizes, int n_in,
                              void* d_out, int out_size, void* d_ws, size_t ws_size,
                              hipStream_t stream) {
  const float* h    = (const float*)d_in[0];
  const float* esh  = (const float*)d_in[1];
  const float* ef   = (const float*)d_in[2];
  const int*   ei   = (const int*)  d_in[3];
  const float* W1   = (const float*)d_in[4];
  const float* b1   = (const float*)d_in[5];
  const float* W2   = (const float*)d_in[6];
  const float* b2   = (const float*)d_in[7];
  const float* W3   = (const float*)d_in[8];
  const float* b3   = (const float*)d_in[9];
  const float* ws_s = (const float*)d_in[10];
  const float* bs_s = (const float*)d_in[11];
  const float* ws_v = (const float*)d_in[12];
  const float* wg_s = (const float*)d_in[13];
  const float* bg_s = (const float*)d_in[14];
  const float* wg_v = (const float*)d_in[15];
  const float* bn_ws= (const float*)d_in[16];
  const float* bn_bs= (const float*)d_in[17];
  const float* bn_wv= (const float*)d_in[18];

  const int N = in_sizes[0] / FDIM;
  const int E = in_sizes[1] / 4;

  // workspace layout (contiguous zero region first)
  float* wsf    = (float*)d_ws;
  float* summed = wsf;                              // N*80
  float* deg    = summed + (size_t)N * FDIM;        // N
  float* stats  = deg + N;                          // 80
  float* upd    = stats + 80;                       // N*80
  float* norm   = upd + (size_t)N * FDIM;           // 80

  // bf16 packed weight fragments, 64B-aligned after the f32 region
  size_t offf = (size_t)N * FDIM + N + 80 + (size_t)N * FDIM + 80;
  offf = (offf + 15) & ~(size_t)15;                 // 64B alignment
  __bf16* pW1 = (__bf16*)(wsf + offf);              // 4 tiles * 1 chunk * 512
  __bf16* pW2 = pW1 + 4 * 512;                      // 4 tiles * 2 chunks * 512
  __bf16* pW3 = pW2 + 8 * 512;                      // 144 tiles * 2 chunks * 512

  const size_t zcount = (size_t)N * FDIM + (size_t)N + 80;
  zero_kernel<<<(int)((zcount + 255) / 256), 256, 0, stream>>>(summed, zcount);

  // one-time weight packing (runs every launch; deterministic)
  pack_b_kernel<<<(2048   + 255) / 256, 256, 0, stream>>>(W1, pW1, HID,  1, 2048);
  pack_b_kernel<<<(4096   + 255) / 256, 256, 0, stream>>>(W2, pW2, HID,  2, 4096);
  pack_b_kernel<<<(147456 + 255) / 256, 256, 0, stream>>>(W3, pW3, WNUM, 2, 147456);

  const int etiles  = (E + 15) / 16;
  const int eblocks = (etiles + 3) / 4;             // 4 waves (tiles) / block
  edge_kernel<<<eblocks, 128, 0, stream>>>(h, esh, ef, ei, pW1, b1, pW2, b2,
                                           pW3, b3, summed, deg, E);

  node_kernel1<<<(N + 127) / 128, 128, 0, stream>>>(h, summed, deg, ws_s, bs_s,
                                                    ws_v, upd, stats, N);
  stats_kernel<<<1, 128, 0, stream>>>(stats, norm, N);
  node_kernel2<<<(N + 127) / 128, 128, 0, stream>>>(h, upd, norm, wg_s, bg_s,
                                                    wg_v, bn_ws, bn_bs, bn_wv,
                                                    (float*)d_out, N);
}
